// DecoderWithAttention_42554535969370
// MI455X (gfx1250) — compile-verified, matrix operand order checked by reading, not run on hardware
//
#include <hip/hip_runtime.h>
#include <hip/hip_bf16.h>
#include <math.h>

#define B_ 128
#define P_ 196
#define MAXLEN_ 30
#define T_ 29
#define E_ 2048
#define D_ 512
#define A_ 512
#define EMB_ 512
#define V_ 7004

typedef __attribute__((ext_vector_type(16))) __bf16 v16bf;
typedef __attribute__((ext_vector_type(8)))  float v8f;
typedef __attribute__((ext_vector_type(4)))  unsigned int v4u;

union FragAB { v16bf v; v4u q[2]; unsigned short u[16]; };
union FragC  { v8f v; float f[8]; };

__device__ __forceinline__ unsigned short f2bf(float f) {
  unsigned int u = __float_as_uint(f);
  unsigned int r = u + 0x7FFFu + ((u >> 16) & 1u);   // round-to-nearest-even
  return (unsigned short)(r >> 16);
}
__device__ __forceinline__ unsigned int pk2bf(float lo, float hi) {
  return (unsigned int)f2bf(lo) | ((unsigned int)f2bf(hi) << 16);
}
__device__ __forceinline__ float sigmoidf_(float x) { return 1.0f / (1.0f + expf(-x)); }

// Map k (0..31) of a 16x32 bf16 operand to (half, element) per the documented
// CDNA5 lane layout: half = (k>>3)&1 ; e = (k&7) + ((k&16)>>1).
__device__ __forceinline__ int khalf(int k) { return (k >> 3) & 1; }
__device__ __forceinline__ int kelem(int k) { return (k & 7) + ((k & 16) >> 1); }

// ---------------------------------------------------------------------------
// Generic tiled GEMM: C[M,N] = post( A[M,K] @ B[K,N] + bias ), f32 in/out,
// bf16 WMMA inner product. Tiles staged into LDS in *fragment order* so each
// lane loads its 32-byte operand with two ds_load_b128.
// Block = 256 threads (8 waves). Tile: BM=128, BN=64, BK=32.
// REQUIRES: M % 128 == 0 and K % 32 == 0 (true for every launch below);
// only N may be ragged.
// ---------------------------------------------------------------------------
__global__ __launch_bounds__(256)
void gemm_bf16_wmma(const float* __restrict__ A, const float* __restrict__ Bm,
                    const float* __restrict__ bias, float* __restrict__ C,
                    int M, int N, int K, int lda, int ldb, int ldc,
                    const int* __restrict__ row_map,
                    const float* __restrict__ row_scale,
                    int act, int accumulate)
{
  // Fragment-major LDS:
  //  A: [row 0..127][half 0..1][e 0..15] bf16  -> 8 KB
  //  B: [j 0..3][lane 0..31][e 0..15]   bf16   -> 4 KB
  __shared__ unsigned short lsAf[128 * 2 * 16];
  __shared__ unsigned short lsBf[4 * 32 * 16];

  const int tid  = threadIdx.x;
  const int wave = tid >> 5;
  const int lane = tid & 31;
  const int half = lane >> 4;
  const int l16  = lane & 15;
  const int m0   = blockIdx.y * 128;
  const int n0   = blockIdx.x * 64;

  // ---- loop-invariant A staging setup: k fixed per thread, 8 rows ----
  const int kA     = (tid & 15) * 2;           // 0,2,..,30
  const int rowA0  = tid >> 4;                 // 0..15 ; rows rowA0 + it*16
  const int aBase  = ((rowA0 * 2 + khalf(kA)) << 4) + kelem(kA);  // +it*512
  const float* apr[8];
#pragma unroll
  for (int it = 0; it < 8; ++it) {
    int gm = m0 + rowA0 + it * 16;             // gm < M guaranteed
    int ar = row_map ? row_map[gm] : gm;
    apr[it] = A + (size_t)ar * lda + kA;
  }

  // ---- loop-invariant B staging setup: n fixed per thread, 4 k-pairs ----
  const int nB    = tid & 63;
  const bool colB = (n0 + nB) < N;
  const float* bpr = Bm + (size_t)((n0 + nB < N) ? (n0 + nB) : (N - 1));
  const int kB0   = (tid >> 6) * 2;            // 0,2,4,6 ; k = kB0 + it*8
  int bIdx[4];
#pragma unroll
  for (int it = 0; it < 4; ++it) {
    int k = kB0 + it * 8;
    int bl = khalf(k) * 16 + (nB & 15);
    bIdx[it] = ((((nB >> 4) * 32) + bl) << 4) + kelem(k);
  }

  FragC acc[4];
#pragma unroll
  for (int j = 0; j < 4; ++j)
#pragma unroll
    for (int i = 0; i < 8; ++i) acc[j].f[i] = 0.0f;

  for (int k0 = 0; k0 < K; k0 += 32) {
    // --- stage A tile: one unconditional global_load_b64 per row ---
#pragma unroll
    for (int it = 0; it < 8; ++it) {
      float2 ld = *(const float2*)(apr[it] + k0);
      *(unsigned int*)&lsAf[aBase + it * 512] = pk2bf(ld.x, ld.y);
    }
    // --- stage B tile: unconditional loads, select-zero for ragged N ---
#pragma unroll
    for (int it = 0; it < 4; ++it) {
      size_t gk = (size_t)(k0 + kB0 + it * 8);
      float f0 = bpr[gk * ldb];
      float f1 = bpr[(gk + 1) * ldb];
      if (!colB) { f0 = 0.0f; f1 = 0.0f; }
      *(unsigned int*)&lsBf[bIdx[it]] = pk2bf(f0, f1);
    }
    __syncthreads();

    // --- A fragment: two ds_load_b128 straight into the operand ---
    FragAB fa;
    {
      const v4u* pa = (const v4u*)&lsAf[((wave * 16 + l16) * 2 + half) << 4];
      fa.q[0] = pa[0];
      fa.q[1] = pa[1];
    }
    // --- 4 B fragments + WMMA ---
#pragma unroll
    for (int j = 0; j < 4; ++j) {
      FragAB fb;
      const v4u* pb = (const v4u*)&lsBf[(j * 32 + lane) << 4];
      fb.q[0] = pb[0];
      fb.q[1] = pb[1];
      acc[j].v = __builtin_amdgcn_wmma_f32_16x16x32_bf16(
          false, fa.v, false, fb.v, (short)0, acc[j].v, false, false);
    }
    __syncthreads();
  }

  // --- epilogue: C VGPR i -> M = i + half*8, N = lane%16 ---
#pragma unroll
  for (int j = 0; j < 4; ++j) {
    int gn = n0 + j * 16 + l16;
#pragma unroll
    for (int i = 0; i < 8; ++i) {
      int gm = m0 + wave * 16 + half * 8 + i;
      if (gn < N) {
        float v = acc[j].f[i];
        if (bias) v += bias[gn];
        if (accumulate) v += C[(size_t)gm * ldc + gn];
        if (act == 1) v = sigmoidf_(v);
        if (row_scale) v *= row_scale[gm];
        C[(size_t)gm * ldc + gn] = v;
      }
    }
  }
}

// ---------------------------------------------------------------------------
// Stable descending argsort of caption lengths (B=128, O(B^2) is trivial),
// plus emission of caps/dec_len/sort_ind outputs (as float per harness).
// ---------------------------------------------------------------------------
__global__ void sort_kernel(const int* __restrict__ cap_len, const int* __restrict__ caps,
                            int* __restrict__ si, int* __restrict__ dl,
                            float* __restrict__ out_caps, float* __restrict__ out_dl,
                            float* __restrict__ out_si)
{
  __shared__ int lens[B_];
  __shared__ int sind[B_];
  int tid = threadIdx.x;
  lens[tid] = cap_len[tid];          // (B,1) flat
  __syncthreads();
  int li = lens[tid];
  int pos = 0;
  for (int j = 0; j < B_; ++j) {
    int lj = lens[j];
    if (lj > li || (lj == li && j < tid)) pos++;
  }
  sind[pos] = tid;
  __syncthreads();
  int s = sind[tid];
  si[tid] = s;
  int d = lens[s] - 1;
  dl[tid] = d;
  out_dl[tid] = (float)d;
  out_si[tid] = (float)s;
  for (int l = 0; l < MAXLEN_; ++l)
    out_caps[tid * MAXLEN_ + l] = (float)caps[s * MAXLEN_ + l];
}

__global__ void rowmap_kernel(const int* __restrict__ si, int* __restrict__ rm) {
  int idx = blockIdx.x * 256 + threadIdx.x;
  if (idx < B_ * P_) {
    int b = idx / P_, p = idx % P_;
    rm[idx] = si[b] * P_ + p;
  }
}

__global__ void meanenc_kernel(const float* __restrict__ enc, const int* __restrict__ si,
                               float* __restrict__ me) {
  int b = blockIdx.x;
  int e = blockIdx.y * 256 + threadIdx.x;
  const float* base = enc + ((size_t)si[b] * P_) * E_ + e;
  float s = 0.0f;
  for (int p = 0; p < P_; ++p) s += base[(size_t)p * E_];
  me[(size_t)b * E_ + e] = s * (1.0f / (float)P_);
}

// ---------------------------------------------------------------------------
// Attention: e = relu(att1 + att2) @ w_fa + b_fa; softmax over P;
// alphas (masked) to d_out; awe = sum_p alpha * enc. One block per batch row.
// ---------------------------------------------------------------------------
__global__ __launch_bounds__(256)
void attention_kernel(const float* __restrict__ att1, const float* __restrict__ att2,
                      const float* __restrict__ wfa, const float* __restrict__ bfa,
                      const float* __restrict__ enc, const int* __restrict__ si,
                      const int* __restrict__ dl,
                      float* __restrict__ awe, float* __restrict__ out_alphas, int t)
{
  __shared__ float att2s[A_];
  __shared__ float wfas[A_];
  __shared__ float esh[P_];
  __shared__ float red[2];
  const int b = blockIdx.x;
  const int tid = threadIdx.x;
  for (int a = tid; a < A_; a += 256) {
    att2s[a] = att2[(size_t)b * A_ + a];
    wfas[a]  = wfa[a];
  }
  __syncthreads();
  const int wid = tid >> 5, lane = tid & 31;
  for (int p = wid; p < P_; p += 8) {
    const float* arow = att1 + ((size_t)b * P_ + p) * A_;
    float s = 0.0f;
    for (int a = lane; a < A_; a += 32) {
      float v = arow[a] + att2s[a];
      v = v > 0.0f ? v : 0.0f;
      s += v * wfas[a];
    }
    for (int m = 16; m > 0; m >>= 1) s += __shfl_xor(s, m, 32);
    if (lane == 0) esh[p] = s + bfa[0];
  }
  __syncthreads();
  if (tid < 32) {
    float mx = -3.4e38f;
    for (int p = tid; p < P_; p += 32) mx = fmaxf(mx, esh[p]);
    for (int m = 16; m > 0; m >>= 1) mx = fmaxf(mx, __shfl_xor(mx, m, 32));
    if (tid == 0) red[0] = mx;
  }
  __syncthreads();
  float mx = red[0];
  for (int p = tid; p < P_; p += 256) esh[p] = expf(esh[p] - mx);
  __syncthreads();
  if (tid < 32) {
    float s = 0.0f;
    for (int p = tid; p < P_; p += 32) s += esh[p];
    for (int m = 16; m > 0; m >>= 1) s += __shfl_xor(s, m, 32);
    if (tid == 0) red[1] = s;
  }
  __syncthreads();
  float inv = 1.0f / red[1];
  float mf = (t < dl[b]) ? 1.0f : 0.0f;
  for (int p = tid; p < P_; p += 256) {
    float a = esh[p] * inv;
    esh[p] = a;
    out_alphas[((size_t)b * T_ + t) * P_ + p] = a * mf;
  }
  __syncthreads();
  const float* encb = enc + ((size_t)si[b] * P_) * E_;
  for (int e = tid; e < E_; e += 256) {
    float s = 0.0f;
    for (int p = 0; p < P_; ++p) s += esh[p] * encb[(size_t)p * E_ + e];
    awe[(size_t)b * E_ + e] = s;
  }
}

__global__ void buildx_kernel(const float* __restrict__ emb, const int* __restrict__ caps,
                              const int* __restrict__ si, const float* __restrict__ gate,
                              const float* __restrict__ awe, float* __restrict__ x, int t)
{
  int b = blockIdx.x;
  int j = blockIdx.y * 256 + threadIdx.x;
  if (j < EMB_) {
    int tok = caps[si[b] * MAXLEN_ + t];
    x[(size_t)b * (EMB_ + E_) + j] = emb[(size_t)tok * EMB_ + j];
  } else if (j < EMB_ + E_) {
    int j2 = j - EMB_;
    x[(size_t)b * (EMB_ + E_) + j] = gate[(size_t)b * E_ + j2] * awe[(size_t)b * E_ + j2];
  }
}

__global__ void lstm_kernel(const float* __restrict__ gates, float* __restrict__ h,
                            float* __restrict__ c, float* __restrict__ hnew,
                            const int* __restrict__ dl, float* __restrict__ mfb, int t)
{
  int b = blockIdx.x;
  int n = blockIdx.y * 256 + threadIdx.x;
  if (n >= D_) return;
  const float* g = gates + (size_t)b * (4 * D_);
  float gi = sigmoidf_(g[n]);
  float gf = sigmoidf_(g[n + D_]);
  float gg = tanhf(g[n + 2 * D_]);
  float go = sigmoidf_(g[n + 3 * D_]);
  float cn = gf * c[(size_t)b * D_ + n] + gi * gg;
  float hn = go * tanhf(cn);
  float mf = (t < dl[b]) ? 1.0f : 0.0f;
  hnew[(size_t)b * D_ + n] = hn;
  if (mf > 0.0f) {
    h[(size_t)b * D_ + n] = hn;
    c[(size_t)b * D_ + n] = cn;
  }
  if (n == 0) mfb[b] = mf;
}

// ---------------------------------------------------------------------------
extern "C" void kernel_launch(void* const* d_in, const int* in_sizes, int n_in,
                              void* d_out, int out_size, void* d_ws, size_t ws_size,
                              hipStream_t stream)
{
  (void)in_sizes; (void)n_in; (void)out_size; (void)ws_size;
  const float* enc  = (const float*)d_in[0];
  const int*  caps  = (const int*)d_in[1];
  const int*  clen  = (const int*)d_in[2];
  const float* emb  = (const float*)d_in[3];
  const float* W_ea = (const float*)d_in[4];
  const float* b_ea = (const float*)d_in[5];
  const float* W_da = (const float*)d_in[6];
  const float* b_da = (const float*)d_in[7];
  const float* w_fa = (const float*)d_in[8];
  const float* b_fa = (const float*)d_in[9];
  const float* W_ih = (const float*)d_in[10];
  const float* b_ih = (const float*)d_in[11];
  const float* W_hh = (const float*)d_in[12];
  const float* b_hh = (const float*)d_in[13];
  const float* W_h0 = (const float*)d_in[14];
  const float* b_h0 = (const float*)d_in[15];
  const float* W_c0 = (const float*)d_in[16];
  const float* b_c0 = (const float*)d_in[17];
  const float* W_fb = (const float*)d_in[18];
  const float* b_fb = (const float*)d_in[19];
  const float* W_fc = (const float*)d_in[20];
  const float* b_fc = (const float*)d_in[21];

  float* out = (float*)d_out;
  const size_t PRED_OFF = 0;
  const size_t CAPS_OFF = (size_t)B_ * T_ * V_;
  const size_t DL_OFF   = CAPS_OFF + (size_t)B_ * MAXLEN_;
  const size_t AL_OFF   = DL_OFF + B_;
  const size_t SI_OFF   = AL_OFF + (size_t)B_ * T_ * P_;

  char* ws = (char*)d_ws;
  size_t off = 0;
  auto alloc = [&](size_t bytes) -> void* {
    void* p = ws + off;
    off = (off + bytes + 255) & ~(size_t)255;
    return p;
  };
  int*   si    = (int*)alloc(B_ * sizeof(int));
  int*   dl    = (int*)alloc(B_ * sizeof(int));
  float* mfb   = (float*)alloc(B_ * sizeof(float));
  int*   rmap  = (int*)alloc((size_t)B_ * P_ * sizeof(int));
  float* me    = (float*)alloc((size_t)B_ * E_ * sizeof(float));
  float* h     = (float*)alloc((size_t)B_ * D_ * sizeof(float));
  float* c     = (float*)alloc((size_t)B_ * D_ * sizeof(float));
  float* hnew  = (float*)alloc((size_t)B_ * D_ * sizeof(float));
  float* att2  = (float*)alloc((size_t)B_ * A_ * sizeof(float));
  float* awe   = (float*)alloc((size_t)B_ * E_ * sizeof(float));
  float* gate  = (float*)alloc((size_t)B_ * E_ * sizeof(float));
  float* x     = (float*)alloc((size_t)B_ * (EMB_ + E_) * sizeof(float));
  float* gates = (float*)alloc((size_t)B_ * 4 * D_ * sizeof(float));
  float* att1  = (float*)alloc((size_t)B_ * P_ * A_ * sizeof(float));  // 51.4 MB

  sort_kernel<<<1, B_, 0, stream>>>(clen, caps, si, dl,
                                    out + CAPS_OFF, out + DL_OFF, out + SI_OFF);
  rowmap_kernel<<<(B_ * P_ + 255) / 256, 256, 0, stream>>>(si, rmap);
  meanenc_kernel<<<dim3(B_, E_ / 256), 256, 0, stream>>>(enc, si, me);

  auto gemm = [&](const float* Am, const float* Bm, const float* bias, float* Cm,
                  int M, int N, int K, int lda, int ldb, int ldc,
                  const int* rm, const float* rs, int act, int accum) {
    dim3 grid((N + 63) / 64, (M + 127) / 128);
    gemm_bf16_wmma<<<grid, 256, 0, stream>>>(Am, Bm, bias, Cm, M, N, K,
                                             lda, ldb, ldc, rm, rs, act, accum);
  };

  // h0 / c0 from mean-pooled encoder features
  gemm(me, W_h0, b_h0, h, B_, D_, E_, E_, D_, D_, nullptr, nullptr, 0, 0);
  gemm(me, W_c0, b_c0, c, B_, D_, E_, E_, D_, D_, nullptr, nullptr, 0, 0);
  // att1 = enc_sorted @ W_ea + b_ea  (row-gathered A, M=25088, K=2048, N=512)
  gemm(enc, W_ea, b_ea, att1, B_ * P_, A_, E_, E_, A_, A_, rmap, nullptr, 0, 0);

  for (int t = 0; t < T_; ++t) {
    gemm(h, W_da, b_da, att2, B_, A_, D_, D_, A_, A_, nullptr, nullptr, 0, 0);
    attention_kernel<<<B_, 256, 0, stream>>>(att1, att2, w_fa, b_fa, enc, si, dl,
                                             awe, out + AL_OFF, t);
    gemm(h, W_fb, b_fb, gate, B_, E_, D_, D_, E_, E_, nullptr, nullptr, 1, 0);  // sigmoid
    buildx_kernel<<<dim3(B_, (EMB_ + E_ + 255) / 256), 256, 0, stream>>>(
        emb, caps, si, gate, awe, x, t);
    gemm(x, W_ih, b_ih, gates, B_, 4 * D_, EMB_ + E_, EMB_ + E_, 4 * D_, 4 * D_,
         nullptr, nullptr, 0, 0);
    gemm(h, W_hh, b_hh, gates, B_, 4 * D_, D_, D_, 4 * D_, 4 * D_,
         nullptr, nullptr, 0, 1);  // accumulate
    lstm_kernel<<<dim3(B_, D_ / 256), 256, 0, stream>>>(gates, h, c, hnew, dl, mfb, t);
    // preds = (h_new @ W_fc + b_fc) * mf, written strided into d_out (ldc = T*V)
    gemm(hnew, W_fc, b_fc, out + PRED_OFF + (size_t)t * V_, B_, V_, D_, D_, V_,
         T_ * V_, nullptr, mfb, 0, 0);
  }
}
